// RefineContextLayer_5119601016852
// MI455X (gfx1250) — matrix operand order repaired and checked by reading, not compiled, not run on hardware
//
#include <hip/hip_runtime.h>
#include <hip/hip_bf16.h>
#include <math.h>

// Problem constants
#define BB 32
#define PP 6
#define DDIM 256
#define HDIM 32
#define SS 3
#define LLEN 512
#define NHEAD 8
#define DFFN 1024
#define MROWS (BB*PP)      // 192
#define SL (SS*LLEN)       // 1536

typedef __bf16 bf16t;
typedef __attribute__((ext_vector_type(16))) __bf16 v16bf;
typedef __attribute__((ext_vector_type(8)))  __bf16 v8bf;
typedef __attribute__((ext_vector_type(8)))  float  v8f;
typedef __attribute__((ext_vector_type(4)))  unsigned int v4u;
typedef __attribute__((ext_vector_type(8)))  int v8i;
typedef __attribute__((ext_vector_type(4)))  int v4i;

// ---- WMMA helpers (CDNA5 16x16x32 bf16, wave32) ----
// A-matrix 16x32 bf16 lane layout (contiguous, matching B): lane l (0-15): M=l, K = 0..15;
// lane l+16: M=l, K = 16..31.  Two contiguous v8bf loads per lane.
__device__ __forceinline__ v16bf load_a(const bf16t* row, int k0, int hf) {
  v8bf lo = *(const v8bf*)(row + k0 + hf * 16);
  v8bf hi = *(const v8bf*)(row + k0 + hf * 16 + 8);
  return __builtin_shufflevector(lo, hi, 0,1,2,3,4,5,6,7,8,9,10,11,12,13,14,15);
}

__device__ __forceinline__ v8f wmma_bf16(v16bf a, v16bf b, v8f c) {
  return __builtin_amdgcn_wmma_f32_16x16x32_bf16(false, a, false, b, (short)0, c, false, false);
}

// ---- transpose + fp32 -> bf16 weight conversion: out[C x R] = (bf16) in[R x C]^T ----
__global__ __launch_bounds__(256) void convT_kernel(const float* __restrict__ in,
                                                    bf16t* __restrict__ out, int R, int C) {
  long long base = (long long)blockIdx.y * R * C;
  long long i = (long long)blockIdx.x * 256 + threadIdx.x;
  if (i >= (long long)R * C) return;
  int r = (int)(i / C), c = (int)(i % C);
  out[base + (long long)c * R + r] = (bf16t)in[base + i];
}

// ---- layernorm rows of width D=256, bf16 output; g/b selected per row group ----
__global__ __launch_bounds__(256) void ln_rows_kernel(const float* __restrict__ x,
                                                      const float* __restrict__ g,
                                                      const float* __restrict__ b,
                                                      bf16t* __restrict__ out,
                                                      int rows_per_g) {
  int row = blockIdx.x, t = threadIdx.x;
  float v = x[(long long)row * DDIM + t];
  __shared__ float s1[DDIM];
  __shared__ float s2[DDIM];
  s1[t] = v; s2[t] = v * v;
  __syncthreads();
  for (int st = 128; st > 0; st >>= 1) {
    if (t < st) { s1[t] += s1[t + st]; s2[t] += s2[t + st]; }
    __syncthreads();
  }
  float mu  = s1[0] * (1.0f / DDIM);
  float var = s2[0] * (1.0f / DDIM) - mu * mu;
  float rs  = rsqrtf(var + 1e-5f);
  int gi = (row / rows_per_g) * DDIM + t;
  out[(long long)row * DDIM + t] = (bf16t)((v - mu) * rs * g[gi] + b[gi]);
}

// ---- WMMA GEMM, 16x64 tile/wave, bf16 out: C = A[MxK] @ Bt[NxK]^T (batched via z) ----
__global__ __launch_bounds__(32) void gemm_bf16_kernel(const bf16t* __restrict__ A,
                                                       const bf16t* __restrict__ Bt,
                                                       bf16t* __restrict__ C,
                                                       int K, int ldc,
                                                       long long bA, long long bB, long long bC) {
  int lane = threadIdx.x;
  int hf = lane >> 4, l16 = lane & 15;
  long long z = blockIdx.z;
  const bf16t* Ar = A + bA * z + (long long)(blockIdx.x * 16 + l16) * K;
  const bf16t* Bg = Bt + bB * z + (long long)(blockIdx.y * 64 + l16) * K;
  v8f acc0 = {}, acc1 = {}, acc2 = {}, acc3 = {};
  for (int k0 = 0; k0 < K; k0 += 32) {
    if (k0 + 32 < K) {
      __builtin_prefetch(Ar + k0 + 32, 0, 3);
      __builtin_prefetch(Bg + k0 + 32, 0, 3);
    }
    v16bf a  = load_a(Ar, k0, hf);
    v16bf b0 = *(const v16bf*)(Bg + 0 * 16 * K + k0 + hf * 16);
    v16bf b1 = *(const v16bf*)(Bg + 1 * 16 * K + k0 + hf * 16);
    v16bf b2 = *(const v16bf*)(Bg + 2 * 16 * K + k0 + hf * 16);
    v16bf b3 = *(const v16bf*)(Bg + 3 * 16 * K + k0 + hf * 16);
    acc0 = wmma_bf16(a, b0, acc0);
    acc1 = wmma_bf16(a, b1, acc1);
    acc2 = wmma_bf16(a, b2, acc2);
    acc3 = wmma_bf16(a, b3, acc3);
  }
  bf16t* Cr = C + bC * z + (long long)(blockIdx.x * 16) * ldc + blockIdx.y * 64;
  #pragma unroll
  for (int i = 0; i < 8; ++i) {
    int m = i + 8 * hf;
    Cr[(long long)m * ldc +  0 + l16] = (bf16t)acc0[i];
    Cr[(long long)m * ldc + 16 + l16] = (bf16t)acc1[i];
    Cr[(long long)m * ldc + 32 + l16] = (bf16t)acc2[i];
    Cr[(long long)m * ldc + 48 + l16] = (bf16t)acc3[i];
  }
}

// ---- WMMA GEMM, 16x64 tile/wave, f32 out with bias + residual ----
__global__ __launch_bounds__(32) void gemm_f32_br_kernel(const bf16t* __restrict__ A,
                                                         const bf16t* __restrict__ Bt,
                                                         const float* __restrict__ bias,
                                                         const float* __restrict__ res,
                                                         float* __restrict__ C,
                                                         int K, int ldc) {
  int lane = threadIdx.x;
  int hf = lane >> 4, l16 = lane & 15;
  const bf16t* Ar = A + (long long)(blockIdx.x * 16 + l16) * K;
  const bf16t* Bg = Bt + (long long)(blockIdx.y * 64 + l16) * K;
  v8f acc[4]; acc[0] = {}; acc[1] = {}; acc[2] = {}; acc[3] = {};
  for (int k0 = 0; k0 < K; k0 += 32) {
    if (k0 + 32 < K) {
      __builtin_prefetch(Ar + k0 + 32, 0, 3);
      __builtin_prefetch(Bg + k0 + 32, 0, 3);
    }
    v16bf a = load_a(Ar, k0, hf);
    #pragma unroll
    for (int s = 0; s < 4; ++s) {
      v16bf b = *(const v16bf*)(Bg + (long long)s * 16 * K + k0 + hf * 16);
      acc[s] = wmma_bf16(a, b, acc[s]);
    }
  }
  #pragma unroll
  for (int s = 0; s < 4; ++s) {
    #pragma unroll
    for (int i = 0; i < 8; ++i) {
      int m = blockIdx.x * 16 + i + 8 * hf;
      int n = blockIdx.y * 64 + s * 16 + l16;
      long long idx = (long long)m * ldc + n;
      C[idx] = acc[s][i] + bias[n] + res[idx];
    }
  }
}

// ---- fused FFN gate, 16x32 N-group (2 sub-tiles x {w1,w3}): G = silu(A@W1+b1)*(A@W3+b3) ----
__global__ __launch_bounds__(32) void gemm_ffn_kernel(const bf16t* __restrict__ A,
                                                      const bf16t* __restrict__ B1t,
                                                      const bf16t* __restrict__ B3t,
                                                      const float* __restrict__ b1,
                                                      const float* __restrict__ b3,
                                                      bf16t* __restrict__ G,
                                                      int K, int ldc) {
  int lane = threadIdx.x;
  int hf = lane >> 4, l16 = lane & 15;
  const bf16t* Ar  = A   + (long long)(blockIdx.x * 16 + l16) * K;
  const bf16t* B1g = B1t + (long long)(blockIdx.y * 32 + l16) * K;
  const bf16t* B3g = B3t + (long long)(blockIdx.y * 32 + l16) * K;
  v8f a1[2]; a1[0] = {}; a1[1] = {};
  v8f a3[2]; a3[0] = {}; a3[1] = {};
  for (int k0 = 0; k0 < K; k0 += 32) {
    v16bf a = load_a(Ar, k0, hf);
    #pragma unroll
    for (int s = 0; s < 2; ++s) {
      v16bf b1v = *(const v16bf*)(B1g + (long long)s * 16 * K + k0 + hf * 16);
      v16bf b3v = *(const v16bf*)(B3g + (long long)s * 16 * K + k0 + hf * 16);
      a1[s] = wmma_bf16(a, b1v, a1[s]);
      a3[s] = wmma_bf16(a, b3v, a3[s]);
    }
  }
  #pragma unroll
  for (int s = 0; s < 2; ++s) {
    #pragma unroll
    for (int i = 0; i < 8; ++i) {
      int m = blockIdx.x * 16 + i + 8 * hf;
      int n = blockIdx.y * 32 + s * 16 + l16;
      float u = a1[s][i] + b1[n];
      float t = a3[s][i] + b3[n];
      float si = u / (1.0f + __expf(-u));   // silu
      G[(long long)m * ldc + n] = (bf16t)(si * t);
    }
  }
}

// ---- fused attention per (b, h): WMMA q.k^T -> mask/scale -> softmax (LDS)
//      -> TDM-staged v tiles in LDS -> WMMA attn.v ----
__global__ __launch_bounds__(32) void attn_kernel(const bf16t* __restrict__ q,     // [B*P, 256] bf16
                                                  const bf16t* __restrict__ kv,    // [S,B,L,512] bf16
                                                  const int*   __restrict__ kvlen, // [S,B]
                                                  bf16t* __restrict__ aout) {      // [B*P, 256] bf16
  int b = blockIdx.x, h = blockIdx.y;
  int lane = threadIdx.x, hf = lane >> 4, l16 = lane & 15;
  __shared__ __align__(32) bf16t sc[16 * SL];        // 48 KB scores / attn weights
  __shared__ __align__(32) bf16t vbuf[128 * HDIM];   // 8 KB TDM-staged v chunk

  int n0 = kvlen[0 * BB + b], n1 = kvlen[1 * BB + b], n2 = kvlen[2 * BB + b];
  float ntot  = (float)(n0 + n1 + n2);
  float scale = logf(ntot + 1.0f) / (sqrtf((float)HDIM) * logf(32.0f));

  // q A-tile: rows p (pad 6->16 with zeros), K = HD = 32
  v16bf aq;
  if (l16 < PP) {
    const bf16t* qr = q + (long long)(b * PP + l16) * DDIM + h * HDIM;
    aq = load_a(qr, 0, hf);
  } else {
    #pragma unroll
    for (int j = 0; j < 16; ++j) aq[j] = (bf16t)0.0f;
  }

  // phase 1: scores = (q @ k^T) * scale, masked, bf16 into LDS
  for (int tile = 0; tile < SL / 16; ++tile) {
    int lg = tile * 16 + l16;
    int s = lg >> 9, l = lg & (LLEN - 1);
    const bf16t* kr = kv + ((((long long)s * BB + b) * LLEN + l) * (2 * DDIM)) + h * (2 * HDIM) + hf * 16;
    v16bf bk = *(const v16bf*)kr;
    v8f sacc = {};
    sacc = wmma_bf16(aq, bk, sacc);
    bool valid = l < kvlen[s * BB + b];
    #pragma unroll
    for (int i = 0; i < 8; ++i) {
      float v = valid ? sacc[i] * scale : -1e10f;
      sc[(i + 8 * hf) * SL + lg] = (bf16t)v;
    }
  }
  __syncthreads();

  // phase 2: row softmax (lane r owns row r), zero pad rows
  if (lane < 16) {
    int r = lane;
    if (r < PP) {
      float mx = -1e30f;
      for (int l = 0; l < SL; ++l) { float v = (float)sc[r * SL + l]; mx = v > mx ? v : mx; }
      float sum = 0.0f;
      for (int l = 0; l < SL; ++l) sum += __expf((float)sc[r * SL + l] - mx);
      float inv = 1.0f / sum;
      for (int l = 0; l < SL; ++l) sc[r * SL + l] = (bf16t)(__expf((float)sc[r * SL + l] - mx) * inv);
    } else {
      for (int l = 0; l < SL; ++l) sc[r * SL + l] = (bf16t)0.0f;
    }
  }
  __syncthreads();

  // phase 3: out = attn @ v ; v chunks (128 rows x 32) staged into LDS by the TDM
  v8f o0 = {}, o1 = {};
  for (int s = 0; s < SS; ++s) {
    for (int chunk = 0; chunk < LLEN / 128; ++chunk) {
      const bf16t* gsrc = kv + ((((long long)s * BB + b) * LLEN + chunk * 128) * (2 * DDIM))
                             + h * (2 * HDIM) + HDIM;
#if __has_builtin(__builtin_amdgcn_tensor_load_to_lds)
      // Build Tensor DMA descriptor (ISA 8.3/8.4): 2D tile 32 x 128, row stride 512 halfs
      unsigned long long ga = (unsigned long long)(const void*)gsrc;
      unsigned lds_off = (unsigned)(unsigned long long)(void*)&vbuf[0];
      v4u g0; v8i g1; v4i g2 = {0,0,0,0}; v4i g3 = {0,0,0,0};
      g0[0] = 1u;                                       // count=1, user mode
      g0[1] = lds_off;                                  // lds_addr (bytes)
      g0[2] = (unsigned)(ga & 0xFFFFFFFFu);             // global_addr[31:0]
      g0[3] = (unsigned)((ga >> 32) & 0x01FFFFFFu) | (2u << 30);  // addr[56:32] | type=2
      g1[0] = (1 << 16);                                // data_size=1 (2 bytes)
      g1[1] = (int)(HDIM << 16);                        // tensor_dim0[15:0] in [31:16]
      g1[2] = (int)(128 << 16);                         // tensor_dim1[15:0] in [31:16]
      g1[3] = (int)(HDIM << 16);                        // tile_dim0 in [31:16]
      g1[4] = 128;                                      // tile_dim1 (tile_dim2=0)
      g1[5] = 2 * DDIM;                                 // tensor_dim0_stride[31:0] = 512
      g1[6] = 0;
      g1[7] = 0;
  #if defined(__clang_major__) && __clang_major__ >= 23
      { v8i g4 = {0,0,0,0,0,0,0,0};
        __builtin_amdgcn_tensor_load_to_lds(g0, g1, g2, g3, g4, 0); }
  #else
      __builtin_amdgcn_tensor_load_to_lds(g0, g1, g2, g3, 0);
  #endif
      __builtin_amdgcn_s_wait_tensorcnt(0);
#else
      for (int r = lane; r < 128; r += 32) {
        const v16bf* src = (const v16bf*)(gsrc + (long long)r * (2 * DDIM));
        *(v16bf*)(vbuf + r * HDIM)      = src[0];
        *(v16bf*)(vbuf + r * HDIM + 16) = src[1];
      }
#endif
      __syncthreads();

      int kbase = s * LLEN + chunk * 128;
      for (int kk = 0; kk < 128; kk += 32) {
        v16bf a = load_a(sc + (long long)l16 * SL, kbase + kk, hf);
        v16bf bv0, bv1;
        #pragma unroll
        for (int j = 0; j < 16; ++j) {
          int r = kk + hf * 16 + j;
          bv0[j] = vbuf[r * HDIM + l16];
          bv1[j] = vbuf[r * HDIM + 16 + l16];
        }
        o0 = wmma_bf16(a, bv0, o0);
        o1 = wmma_bf16(a, bv1, o1);
      }
      __syncthreads();
    }
  }
  #pragma unroll
  for (int i = 0; i < 8; ++i) {
    int p = i + 8 * hf;
    if (p < PP) {
      long long base = (long long)(b * PP + p) * DDIM + h * HDIM;
      aout[base + l16]      = (bf16t)o0[i];
      aout[base + 16 + l16] = (bf16t)o1[i];
    }
  }
}

extern "C" void kernel_launch(void* const* d_in, const int* in_sizes, int n_in,
                              void* d_out, int out_size, void* d_ws, size_t ws_size,
                              hipStream_t stream) {
  const float* raw_q    = (const float*)d_in[0];
  const float* kv_pad   = (const float*)d_in[1];
  const int*   kv_len   = (const int*)  d_in[2];
  const float* wq_ln_g  = (const float*)d_in[3];
  const float* wq_ln_b  = (const float*)d_in[4];
  const float* wq_w     = (const float*)d_in[5];
  const float* wkv_ln_g = (const float*)d_in[6];
  const float* wkv_ln_b = (const float*)d_in[7];
  const float* wkv_w    = (const float*)d_in[8];
  const float* fc_w     = (const float*)d_in[9];
  const float* fc_b     = (const float*)d_in[10];
  const float* ffn_ln_g = (const float*)d_in[11];
  const float* ffn_ln_b = (const float*)d_in[12];
  const float* w1_w     = (const float*)d_in[13];
  const float* w1_b     = (const float*)d_in[14];
  const float* w2_w     = (const float*)d_in[15];
  const float* w2_b     = (const float*)d_in[16];
  const float* w3_w     = (const float*)d_in[17];
  const float* w3_b     = (const float*)d_in[18];
  float* outp = (float*)d_out;
  (void)in_sizes; (void)n_in; (void)out_size;

  size_t off = 0;
  auto alloc = [&](size_t bytes) { size_t o = off; off += (bytes + 255) & ~(size_t)255; return o; };
  size_t o_qln  = alloc((size_t)MROWS * DDIM * 2);
  size_t o_kvn  = alloc((size_t)SS * BB * LLEN * DDIM * 2);
  size_t o_wqT  = alloc((size_t)DDIM * DDIM * 2);
  size_t o_wkvT = alloc((size_t)SS * (2 * DDIM) * DDIM * 2);
  size_t o_fcT  = alloc((size_t)DDIM * DDIM * 2);
  size_t o_w1T  = alloc((size_t)DFFN * DDIM * 2);
  size_t o_w3T  = alloc((size_t)DFFN * DDIM * 2);
  size_t o_w2T  = alloc((size_t)DDIM * DFFN * 2);
  size_t o_q    = alloc((size_t)MROWS * DDIM * 2);
  size_t o_kvo  = alloc((size_t)SS * BB * LLEN * (2 * DDIM) * 2);
  size_t o_ao   = alloc((size_t)MROWS * DDIM * 2);
  size_t o_x    = alloc((size_t)MROWS * DDIM * 4);
  size_t o_hln  = alloc((size_t)MROWS * DDIM * 2);
  size_t o_g    = alloc((size_t)MROWS * DFFN * 2);
  if (off > ws_size) return;

  char* ws = (char*)d_ws;
  bf16t* qln  = (bf16t*)(ws + o_qln);
  bf16t* kvn  = (bf16t*)(ws + o_kvn);
  bf16t* wqT  = (bf16t*)(ws + o_wqT);
  bf16t* wkvT = (bf16t*)(ws + o_wkvT);
  bf16t* fcT  = (bf16t*)(ws + o_fcT);
  bf16t* w1T  = (bf16t*)(ws + o_w1T);
  bf16t* w3T  = (bf16t*)(ws + o_w3T);
  bf16t* w2T  = (bf16t*)(ws + o_w2T);
  bf16t* qb   = (bf16t*)(ws + o_q);
  bf16t* kvo  = (bf16t*)(ws + o_kvo);
  bf16t* ao   = (bf16t*)(ws + o_ao);
  float* xbuf = (float*)(ws + o_x);
  bf16t* hln  = (bf16t*)(ws + o_hln);
  bf16t* gbuf = (bf16t*)(ws + o_g);

  // 1) weight transpose + bf16 conversion
  convT_kernel<<<dim3(DDIM * DDIM / 256, 1), 256, 0, stream>>>(wq_w, wqT, DDIM, DDIM);
  convT_kernel<<<dim3(DDIM * 2 * DDIM / 256, SS), 256, 0, stream>>>(wkv_w, wkvT, DDIM, 2 * DDIM);
  convT_kernel<<<dim3(DDIM * DDIM / 256, 1), 256, 0, stream>>>(fc_w, fcT, DDIM, DDIM);
  convT_kernel<<<dim3(DDIM * DFFN / 256, 1), 256, 0, stream>>>(w1_w, w1T, DDIM, DFFN);
  convT_kernel<<<dim3(DDIM * DFFN / 256, 1), 256, 0, stream>>>(w3_w, w3T, DDIM, DFFN);
  convT_kernel<<<dim3(DFFN * DDIM / 256, 1), 256, 0, stream>>>(w2_w, w2T, DFFN, DDIM);

  // 2) layernorms -> bf16
  ln_rows_kernel<<<MROWS, 256, 0, stream>>>(raw_q, wq_ln_g, wq_ln_b, qln, MROWS);
  ln_rows_kernel<<<SS * BB * LLEN, 256, 0, stream>>>(kv_pad, wkv_ln_g, wkv_ln_b, kvn, BB * LLEN);

  // 3) q = LN(q) @ wq : (192x256)@(256x256), 16x64 tiles
  gemm_bf16_kernel<<<dim3(MROWS / 16, DDIM / 64, 1), 32, 0, stream>>>(
      qln, wqT, qb, DDIM, DDIM, 0, 0, 0);

  // 4) kv = LN(kv) @ wkv : per s (16384x256)@(256x512), batched over z
  gemm_bf16_kernel<<<dim3(BB * LLEN / 16, 2 * DDIM / 64, SS), 32, 0, stream>>>(
      kvn, wkvT, kvo, DDIM, 2 * DDIM,
      (long long)BB * LLEN * DDIM, (long long)(2 * DDIM) * DDIM,
      (long long)BB * LLEN * (2 * DDIM));

  // 5) attention per (b, h)
  attn_kernel<<<dim3(BB, NHEAD), 32, 0, stream>>>(qb, kvo, kv_len, ao);

  // 6) x = attn_out @ fc_w + fc_b + raw_q (f32)
  gemm_f32_br_kernel<<<dim3(MROWS / 16, DDIM / 64), 32, 0, stream>>>(
      ao, fcT, fc_b, raw_q, xbuf, DDIM, DDIM);

  // 7) h = LN(x)
  ln_rows_kernel<<<MROWS, 256, 0, stream>>>(xbuf, ffn_ln_g, ffn_ln_b, hln, MROWS);

  // 8) g = silu(h@w1+b1) * (h@w3+b3)
  gemm_ffn_kernel<<<dim3(MROWS / 16, DFFN / 32), 32, 0, stream>>>(
      hln, w1T, w3T, w1_b, w3_b, gbuf, DDIM, DFFN);

  // 9) out = x + g @ w2 + b2
  gemm_f32_br_kernel<<<dim3(MROWS / 16, DDIM / 64), 32, 0, stream>>>(
      gbuf, w2T, w2_b, xbuf, outp, DFFN, DDIM);
}
